// LinearAttention1D_76312978915908
// MI455X (gfx1250) — matrix operand order repaired and checked by reading, not compiled, not run on hardware
//
#include <hip/hip_runtime.h>
#include <hip/hip_bf16.h>
#include <math.h>

#define B_     16
#define C_     128
#define T_     16384
#define O_     384
#define HEADS_ 4
#define DHEAD_ 32
#define GN_EPS_ 1e-5f

typedef __attribute__((ext_vector_type(16))) __bf16 v16bf;
typedef __attribute__((ext_vector_type(8)))  float  v8f;
typedef __attribute__((ext_vector_type(4)))  int    v4i;

#define GLOBAL_AS __attribute__((address_space(1)))
#define LDS_AS    __attribute__((address_space(3)))

#if defined(__gfx1250__) && __has_builtin(__builtin_amdgcn_global_load_async_to_lds_b128) && __has_builtin(__builtin_amdgcn_s_wait_asynccnt)
#define ASYNC_COPY 1
#else
#define ASYNC_COPY 0
#endif

static __device__ __forceinline__ __bf16 to_bf16(float f) {
  unsigned u = __builtin_bit_cast(unsigned, f);
  unsigned r = u + 0x7FFFu + ((u >> 16) & 1u);   // round-to-nearest-even
  unsigned short h = (unsigned short)(r >> 16);
  return __builtin_bit_cast(__bf16, h);
}
static __device__ __forceinline__ float bfu_lo(unsigned w) {
  return __builtin_bit_cast(float, w << 16);
}
static __device__ __forceinline__ float bfu_hi(unsigned w) {
  return __builtin_bit_cast(float, w & 0xFFFF0000u);
}
static __device__ __forceinline__ void unpack8(uint4 v, float* f) {
  f[0] = bfu_lo(v.x); f[1] = bfu_hi(v.x); f[2] = bfu_lo(v.y); f[3] = bfu_hi(v.y);
  f[4] = bfu_lo(v.z); f[5] = bfu_hi(v.z); f[6] = bfu_lo(v.w); f[7] = bfu_hi(v.w);
}

// ---------------------------------------------------------------------------
// Kernel 1: qkv[b,o,n] = sum_c W[o,c] * x[b,c,n]   (M=384, K=128, N=16384)
// 128x128 tile, bf16 WMMA, fp32 accumulate; bf16 output via LDS for b128 stores.
// ---------------------------------------------------------------------------
__global__ __launch_bounds__(256)
void qkv_gemm_kernel(const float* __restrict__ x, const float* __restrict__ w,
                     __bf16* __restrict__ qkv) {
  __shared__ __bf16 As[128][136];   // [m][k]; reused as Out[m][n]
  __shared__ __bf16 Bs[128][136];   // [n][k]
  const int tid = threadIdx.x;
  const int n0  = blockIdx.x * 128;
  const int m0  = blockIdx.y * 128;
  const int b   = blockIdx.z;

  #pragma unroll
  for (int i = 0; i < 16; ++i) {                    // A tile (W rows)
    const int f = tid + i * 256;
    const int r = f >> 5, c = (f & 31) * 4;
    float4 v = *(const float4*)(w + (size_t)(m0 + r) * C_ + c);
    As[r][c + 0] = to_bf16(v.x); As[r][c + 1] = to_bf16(v.y);
    As[r][c + 2] = to_bf16(v.z); As[r][c + 3] = to_bf16(v.w);
  }
  #pragma unroll
  for (int i = 0; i < 16; ++i) {                    // B tile (x), transposed [n][c]
    const int f = tid + i * 256;
    const int c = f >> 5, n = (f & 31) * 4;
    float4 v = *(const float4*)(x + ((size_t)b * C_ + c) * T_ + n0 + n);
    Bs[n + 0][c] = to_bf16(v.x); Bs[n + 1][c] = to_bf16(v.y);
    Bs[n + 2][c] = to_bf16(v.z); Bs[n + 3][c] = to_bf16(v.w);
  }
  __syncthreads();

  const int wv = tid >> 5, lane = tid & 31;
  const int lm = lane & 15, grp = lane >> 4;
  const int mbase = wv * 16;

  v8f acc[8];
  #pragma unroll
  for (int j = 0; j < 8; ++j) acc[j] = (v8f){0.f,0.f,0.f,0.f,0.f,0.f,0.f,0.f};

  #pragma unroll
  for (int ks = 0; ks < 4; ++ks) {
    const int kk = ks * 32;
    v16bf a;
    #pragma unroll
    for (int i = 0; i < 8; ++i) {                   // ISA 16-bit A layout
      a[i]     = As[mbase + lm][kk + grp * 8 + i];
      a[i + 8] = As[mbase + lm][kk + 16 + grp * 8 + i];
    }
    #pragma unroll
    for (int j = 0; j < 8; ++j) {
      v16bf bb;
      #pragma unroll
      for (int i = 0; i < 8; ++i) {                 // ISA 16-bit B layout
        bb[i]     = Bs[j * 16 + lm][kk + grp * 8 + i];
        bb[i + 8] = Bs[j * 16 + lm][kk + 16 + grp * 8 + i];
      }
      acc[j] = __builtin_amdgcn_wmma_f32_16x16x32_bf16(
          false, a, false, bb, (short)0, acc[j], false, false);
    }
  }
  __syncthreads();                                  // all WMMA reads of As done
  #pragma unroll
  for (int j = 0; j < 8; ++j)
    #pragma unroll
    for (int r = 0; r < 8; ++r)
      As[mbase + r + 8 * grp][j * 16 + lm] = to_bf16(acc[j][r]);
  __syncthreads();
  #pragma unroll
  for (int i = 0; i < 8; ++i) {                     // coalesced b128 bf16 stores
    const int f = tid + i * 256;                    // 2048 x uint4
    const int r = f >> 4, cc = (f & 15) * 8;
    uint4 o = *(const uint4*)&As[r][cc];
    *(uint4*)(qkv + ((size_t)b * O_ + m0 + r) * T_ + n0 + cc) = o;
  }
}

// ---------------------------------------------------------------------------
// Kernel 2: per-row softmax stats (max, 1/sum_exp); rows 0..2047=q, 2048..=k.
// ---------------------------------------------------------------------------
__global__ __launch_bounds__(256)
void softmax_stats_kernel(const __bf16* __restrict__ qkv, float2* __restrict__ stats) {
  const int row = blockIdx.x;
  const int tid = threadIdx.x;
  const int isK = row >= (B_ * C_);
  const int r2  = isK ? row - B_ * C_ : row;
  const int b = r2 / C_, ch = r2 % C_;
  const __bf16* p = qkv + ((size_t)b * O_ + (isK ? C_ : 0) + ch) * T_;
  __shared__ float red[256];

  float m = -INFINITY;
  for (int i = tid; i < T_ / 8; i += 256) {
    float f[8]; unpack8(((const uint4*)p)[i], f);
    #pragma unroll
    for (int j = 0; j < 8; ++j) m = fmaxf(m, f[j]);
  }
  red[tid] = m; __syncthreads();
  for (int s = 128; s > 0; s >>= 1) {
    if (tid < s) red[tid] = fmaxf(red[tid], red[tid + s]);
    __syncthreads();
  }
  const float rmax = red[0];
  __syncthreads();

  float sum = 0.f;
  for (int i = tid; i < T_ / 8; i += 256) {
    float f[8]; unpack8(((const uint4*)p)[i], f);
    #pragma unroll
    for (int j = 0; j < 8; ++j) sum += __expf(f[j] - rmax);
  }
  red[tid] = sum; __syncthreads();
  for (int s = 128; s > 0; s >>= 1) {
    if (tid < s) red[tid] += red[tid + s];
    __syncthreads();
  }
  if (tid == 0) stats[row] = make_float2(rmax, 1.0f / red[0]);
}

// ---------------------------------------------------------------------------
__global__ void zero_kernel(float* __restrict__ p, int n) {
  int i = blockIdx.x * blockDim.x + threadIdx.x;
  if (i < n) p[i] = 0.f;
}

// ---------------------------------------------------------------------------
// Kernel 3: ctx_raw[b,h,d,e] += sum_n exp(k[d,n]-kmax[d]) * v[e,n]
// v-tiles staged with async global->LDS copies (double buffered) when available.
// ---------------------------------------------------------------------------
__global__ __launch_bounds__(256)
void context_kernel(const __bf16* __restrict__ qkv, const float2* __restrict__ stats,
                    float* __restrict__ ctx) {
  const int chunk = blockIdx.x;   // 0..7 -> 2048 cols each
  const int h = blockIdx.y, b = blockIdx.z;
  const int tid = threadIdx.x;
  __shared__ float ks[32][68];
  __shared__ float vs[32][68];
#if ASYNC_COPY
  __shared__ __bf16 vraw[2][32][64];
#endif

  const int n0 = chunk * 2048;
  const __bf16* kbase = qkv + ((size_t)b * O_ + C_ + h * DHEAD_) * T_;
  const __bf16* vbase = qkv + ((size_t)b * O_ + 2 * C_ + h * DHEAD_) * T_;

  const int d  = tid & 31;        // output row owned by this thread
  const int eb = tid >> 5;        // 0..7 -> e = eb + 8j
  const int lr = tid >> 3;        // load row 0..31
  const int lc = (tid & 7) * 8;   // load col base (8 bf16 = 16B)
  const float kmax = stats[2048 + b * C_ + h * DHEAD_ + lr].x;

#if ASYNC_COPY
  __builtin_amdgcn_global_load_async_to_lds_b128(
      (GLOBAL_AS v4i*)(__bf16*)(vbase + (size_t)lr * T_ + n0 + lc),
      (LDS_AS v4i*)&vraw[0][lr][lc], 0, 0);
#endif

  float acc[4] = {0.f, 0.f, 0.f, 0.f};
  for (int s = 0; s < 2048; s += 64) {
    uint4 kq = *(const uint4*)(kbase + (size_t)lr * T_ + n0 + s + lc);
#if ASYNC_COPY
    const int buf = (s >> 6) & 1;
    if (s + 64 < 2048)
      __builtin_amdgcn_global_load_async_to_lds_b128(
          (GLOBAL_AS v4i*)(__bf16*)(vbase + (size_t)lr * T_ + n0 + s + 64 + lc),
          (LDS_AS v4i*)&vraw[buf ^ 1][lr][lc], 0, 0);
#else
    uint4 vq = *(const uint4*)(vbase + (size_t)lr * T_ + n0 + s + lc);
#endif
    __syncthreads();              // previous sub-tile fully consumed
    {
      float f[8]; unpack8(kq, f);
      #pragma unroll
      for (int j = 0; j < 8; ++j) ks[lr][lc + j] = __expf(f[j] - kmax);
    }
#if ASYNC_COPY
    if (s + 64 < 2048) __builtin_amdgcn_s_wait_asynccnt(1);
    else               __builtin_amdgcn_s_wait_asynccnt(0);
    uint4 vq = *(const uint4*)&vraw[buf][lr][lc];   // own region: own-wave wait ok
#endif
    {
      float f[8]; unpack8(vq, f);
      #pragma unroll
      for (int j = 0; j < 8; ++j) vs[lr][lc + j] = f[j];
    }
    __syncthreads();
    #pragma unroll 16
    for (int n = 0; n < 64; ++n) {
      const float kd = ks[d][n];
      #pragma unroll
      for (int j = 0; j < 4; ++j) acc[j] += kd * vs[eb + 8 * j][n];
    }
  }
  #pragma unroll
  for (int j = 0; j < 4; ++j)
    atomicAdd(&ctx[(((size_t)b * HEADS_ + h) * DHEAD_ + d) * DHEAD_ + eb + 8 * j],
              acc[j]);
}

// ---------------------------------------------------------------------------
// Kernel 4 (fused): softmax(q) -> attn = ctx^T x q (WMMA) -> y = W_out x attn
// + bias (WMMA) -> GN partial sums. attn never leaves LDS; y stored bf16.
// ---------------------------------------------------------------------------
__global__ __launch_bounds__(256)
void attn_out_kernel(const __bf16* __restrict__ qkv, const float2* __restrict__ stats,
                     const float* __restrict__ ctx, const float* __restrict__ w_out,
                     const float* __restrict__ bias, __bf16* __restrict__ y,
                     float* __restrict__ bsum, float* __restrict__ bss) {
  __shared__ __bf16 Wl[128][136];            // W_out [c_out][c_in]
  __shared__ __bf16 qs[128][136];            // softmax(q), [n][d_global]
  __shared__ __bf16 at[128][136];            // attn [n][c_in]; reused as Y[c][n]
  __shared__ float  ctxl[HEADS_][32][33];    // scaled context [h][d][e]
  __shared__ float  red[256], red2[256];
  const int tid = threadIdx.x;
  const int n0  = blockIdx.x * 128;
  const int b   = blockIdx.y;

  #pragma unroll
  for (int i = 0; i < 16; ++i) {             // W_out tile
    const int f = tid + i * 256;
    const int r = f >> 5, c = (f & 31) * 4;
    float4 v = *(const float4*)(w_out + (size_t)r * C_ + c);
    Wl[r][c + 0] = to_bf16(v.x); Wl[r][c + 1] = to_bf16(v.y);
    Wl[r][c + 2] = to_bf16(v.z); Wl[r][c + 3] = to_bf16(v.w);
  }
  for (int i = tid; i < HEADS_ * 32 * 32; i += 256) {  // ctx scaled by k inv-sum
    const int h = i >> 10, dd = (i >> 5) & 31, e = i & 31;
    const float kinv = stats[2048 + b * C_ + h * 32 + dd].y;
    ctxl[h][dd][e] = ctx[(size_t)b * (HEADS_ * 32 * 32) + i] * kinv;
  }
  #pragma unroll
  for (int i = 0; i < 8; ++i) {              // softmax(q) tile -> [n][ch]
    const int f = tid + i * 256;             // 2048 x uint4 over 128x128 bf16
    const int ch = f >> 4, n = (f & 15) * 8;
    const float2 st = stats[b * C_ + ch];
    const float sc = st.y * 0.17677669529663687f;  // inv_sum * dim_head^-0.5
    float fq[8]; unpack8(*(const uint4*)(qkv + ((size_t)b * O_ + ch) * T_ + n0 + n), fq);
    #pragma unroll
    for (int j = 0; j < 8; ++j)
      qs[n + j][ch] = to_bf16(__expf(fq[j] - st.x) * sc);
  }
  __syncthreads();

  const int wv = tid >> 5, lane = tid & 31;
  const int lm = lane & 15, grp = lane >> 4;

  // Stage A: attn[e_glob, n] = sum_d ctx[d,e]*q[d,n]; wave -> (head, e-halftile)
  {
    const int h  = wv >> 1;
    const int mt = (wv & 1) * 16;
    v16bf a;
    #pragma unroll
    for (int i = 0; i < 8; ++i) {
      a[i]     = to_bf16(ctxl[h][grp * 8 + i][mt + lm]);
      a[i + 8] = to_bf16(ctxl[h][16 + grp * 8 + i][mt + lm]);
    }
    #pragma unroll
    for (int j = 0; j < 8; ++j) {
      v16bf bb;
      #pragma unroll
      for (int i = 0; i < 8; ++i) {
        bb[i]     = qs[j * 16 + lm][h * 32 + grp * 8 + i];
        bb[i + 8] = qs[j * 16 + lm][h * 32 + 16 + grp * 8 + i];
      }
      v8f c0 = (v8f){0.f,0.f,0.f,0.f,0.f,0.f,0.f,0.f};
      c0 = __builtin_amdgcn_wmma_f32_16x16x32_bf16(
          false, a, false, bb, (short)0, c0, false, false);
      const int n = j * 16 + lm;
      #pragma unroll
      for (int r = 0; r < 8; ++r)
        at[n][h * 32 + mt + r + 8 * grp] = to_bf16(c0[r]);
    }
  }
  __syncthreads();

  // Stage B: Y = W_out x attn + bias
  const int mbase = wv * 16;
  v8f acc[8];
  #pragma unroll
  for (int j = 0; j < 8; ++j) acc[j] = (v8f){0.f,0.f,0.f,0.f,0.f,0.f,0.f,0.f};
  #pragma unroll
  for (int ks = 0; ks < 4; ++ks) {
    const int kk = ks * 32;
    v16bf a;
    #pragma unroll
    for (int i = 0; i < 8; ++i) {
      a[i]     = Wl[mbase + lm][kk + grp * 8 + i];
      a[i + 8] = Wl[mbase + lm][kk + 16 + grp * 8 + i];
    }
    #pragma unroll
    for (int j = 0; j < 8; ++j) {
      v16bf bb;
      #pragma unroll
      for (int i = 0; i < 8; ++i) {
        bb[i]     = at[j * 16 + lm][kk + grp * 8 + i];
        bb[i + 8] = at[j * 16 + lm][kk + 16 + grp * 8 + i];
      }
      acc[j] = __builtin_amdgcn_wmma_f32_16x16x32_bf16(
          false, a, false, bb, (short)0, acc[j], false, false);
    }
  }
  float bv[8];
  #pragma unroll
  for (int r = 0; r < 8; ++r) bv[r] = bias[mbase + r + 8 * grp];

  __syncthreads();                            // everyone done reading at[] as attn
  #pragma unroll
  for (int j = 0; j < 8; ++j)
    #pragma unroll
    for (int r = 0; r < 8; ++r)
      at[mbase + r + 8 * grp][j * 16 + lm] = to_bf16(acc[j][r] + bv[r]);
  __syncthreads();

  float lsum = 0.f, lss = 0.f;                // copy out + GN partial sums
  #pragma unroll
  for (int i = 0; i < 8; ++i) {
    const int f = tid + i * 256;
    const int r = f >> 4, cc = (f & 15) * 8;
    uint4 o = *(const uint4*)&at[r][cc];
    *(uint4*)(y + ((size_t)b * C_ + r) * T_ + n0 + cc) = o;
    float fv[8]; unpack8(o, fv);
    #pragma unroll
    for (int j = 0; j < 8; ++j) { lsum += fv[j]; lss += fv[j] * fv[j]; }
  }
  red[tid] = lsum; red2[tid] = lss; __syncthreads();
  for (int s = 128; s > 0; s >>= 1) {
    if (tid < s) { red[tid] += red[tid + s]; red2[tid] += red2[tid + s]; }
    __syncthreads();
  }
  if (tid == 0) { atomicAdd(&bsum[b], red[0]); atomicAdd(&bss[b], red2[0]); }
}

// ---------------------------------------------------------------------------
// Kernel 5: GroupNorm finalize: out = (y-mean)*rsqrt(var+eps)*g[ch] + b[ch]
// ---------------------------------------------------------------------------
__global__ __launch_bounds__(256)
void gn_finalize_kernel(const __bf16* __restrict__ y, const float* __restrict__ bsum,
                        const float* __restrict__ bss, const float* __restrict__ gn_w,
                        const float* __restrict__ gn_b, float* __restrict__ out) {
  const size_t i8 = (size_t)blockIdx.x * 256 + threadIdx.x;  // 8-elem chunks
  const size_t elem = i8 * 8;
  const int b  = (int)(elem / ((size_t)C_ * T_));
  const int ch = (int)((elem / T_) % C_);
  const float invN = 1.0f / (float)(C_ * T_);
  const float mean = bsum[b] * invN;
  const float var  = bss[b] * invN - mean * mean;
  const float rs   = rsqrtf(var + GN_EPS_);
  const float g = gn_w[ch] * rs, bb = gn_b[ch];
  float f[8]; unpack8(((const uint4*)y)[i8], f);
  float4 o0, o1;
  o0.x = (f[0] - mean) * g + bb; o0.y = (f[1] - mean) * g + bb;
  o0.z = (f[2] - mean) * g + bb; o0.w = (f[3] - mean) * g + bb;
  o1.x = (f[4] - mean) * g + bb; o1.y = (f[5] - mean) * g + bb;
  o1.z = (f[6] - mean) * g + bb; o1.w = (f[7] - mean) * g + bb;
  ((float4*)out)[i8 * 2 + 0] = o0;
  ((float4*)out)[i8 * 2 + 1] = o1;
}

// ---------------------------------------------------------------------------
extern "C" void kernel_launch(void* const* d_in, const int* in_sizes, int n_in,
                              void* d_out, int out_size, void* d_ws, size_t ws_size,
                              hipStream_t stream) {
  const float* x     = (const float*)d_in[0];
  const float* w_qkv = (const float*)d_in[1];
  const float* w_out = (const float*)d_in[2];
  const float* b_out = (const float*)d_in[3];
  const float* gn_w  = (const float*)d_in[4];
  const float* gn_b  = (const float*)d_in[5];
  float* out = (float*)d_out;

  char* ws = (char*)d_ws;
  __bf16* qkv   = (__bf16*)(ws);                      // 201,326,592 B
  __bf16* y     = (__bf16*)(ws + 201326592ull);       //  67,108,864 B
  float2* stats = (float2*)(ws + 268435456ull);       //      32,768 B
  float*  ctx   = (float*) (ws + 268468224ull);       //     262,144 B
  float*  bsum  = (float*) (ws + 268730368ull);       //          128 B (bsum|bss)
  float*  bss   = bsum + B_;

  // ctx + bsum + bss contiguous: re-zero every call (atomically accumulated).
  zero_kernel<<<dim3((65536 + 32 + 255) / 256), 256, 0, stream>>>(ctx, 65536 + 32);

  qkv_gemm_kernel<<<dim3(T_ / 128, O_ / 128, B_), 256, 0, stream>>>(x, w_qkv, qkv);
  softmax_stats_kernel<<<dim3(2 * B_ * C_), 256, 0, stream>>>(qkv, stats);
  context_kernel<<<dim3(8, HEADS_, B_), 256, 0, stream>>>(qkv, stats, ctx);
  attn_out_kernel<<<dim3(T_ / 128, B_), 256, 0, stream>>>(qkv, stats, ctx, w_out,
                                                          b_out, y, bsum, bss);
  gn_finalize_kernel<<<dim3(B_ * C_ * T_ / 8 / 256), 256, 0, stream>>>(
      y, bsum, bss, gn_w, gn_b, out);
}
